// CCNet_26697516712682
// MI455X (gfx1250) — matrix operand order repaired and checked by reading, not compiled, not run on hardware
//
#include <hip/hip_runtime.h>
#include <hip/hip_bf16.h>

// ---------------------------------------------------------------------------
// CDNA5 (gfx1250) implementation of the CCNet set-transformer reference.
//  - All dense projections use v_wmma_f32_16x16x32_bf16 (bf16 in, f32 acc),
//    64x16 C tile per wave (4 WMMAs / K-step, B fragment reused 4x).
//  - Weights are pre-packed per GEMM into the WMMA B-fragment layout (bf16),
//    so the hot loop's B access is one contiguous 32B load per lane.
//  - Attention over 6714 embedding keys uses an online-softmax VALU kernel.
//  - Everything else (LN, gather, masks) is bandwidth-trivial.
// ---------------------------------------------------------------------------

typedef __bf16 v16bf __attribute__((ext_vector_type(16)));
typedef float  v8f   __attribute__((ext_vector_type(8)));

union U16bf { v16bf v; __bf16 e[16]; };
union U8f   { v8f   v; float  e[8]; };

#define GF_BIAS 1
#define GF_RELU 2
#define GF_RES  4

// ----------------------------------------------------------------------------
// Weight pre-pack: W[K,N] f32 -> P[nTile][kBlock][lane][16] bf16, matching the
// B-fragment lane layout (lanes 0-15: N=lane, K=kb+j; lanes 16-31: K=kb+16+j).
// grid = (nTiles, K/32), block = 32.  Reads coalesced, writes contiguous.
// ----------------------------------------------------------------------------
__global__ __launch_bounds__(32)
void packW_kernel(const float* __restrict__ W, __bf16* __restrict__ P, int N)
{
    const int lane = threadIdx.x & 31;
    const int half = lane >> 4;
    const int l16  = lane & 15;
    const int nT   = blockIdx.x;
    const int kb   = blockIdx.y << 5;
    int nC = (nT << 4) + l16;
    if (nC >= N) nC = N - 1;
    __bf16* dst = P + (((size_t)nT * gridDim.y + blockIdx.y) * 32 + lane) * 16;
#pragma unroll
    for (int j = 0; j < 16; ++j)
        dst[j] = (__bf16)W[(size_t)(kb + j + (half << 4)) * N + nC];
}

// ----------------------------------------------------------------------------
// WMMA GEMM: C[M,N] = A[M,K](lda) @ W[K,N] (+bias)(+relu)(+residual)
// One wave per 64x16 tile of C (4 stacked 16x16 WMMA tiles); K stepped by 32.
// A-fragment lane layout (16-bit A 16x32), same contiguous k order as packed B:
//   lanes 0-15 hold M=lane; element j -> K = j + (lane>=16?16:0)
//   => per lane, elements are one contiguous 16-float run -> 4x b128 loads.
// B-fragment comes pre-packed in bf16: one 32-byte contiguous load per lane
// per K-step, reused by 4 WMMAs.
// C/D layout: VGPR r -> M = r + 8*(lane>=16), N = lane&15.
// OOB: load indices clamped (no exec divergence), stores bounds-guarded.
// ----------------------------------------------------------------------------
__global__ __launch_bounds__(32)
void gemm_bf16_wmma(const float* __restrict__ A, int lda,
                    const __bf16* __restrict__ Wp,
                    const float* __restrict__ bias,
                    const float* __restrict__ res,
                    float* __restrict__ C,
                    int M, int N, int K, int flags)
{
    const int lane = threadIdx.x & 31;
    const int half = lane >> 4;     // 0 or 1
    const int l16  = lane & 15;
    const int n0   = blockIdx.x << 4;
    const int m0   = blockIdx.y << 6;   // 64 rows of C per wave

    U8f acc[4];
#pragma unroll
    for (int t = 0; t < 4; ++t)
#pragma unroll
        for (int r = 0; r < 8; ++r) acc[t].e[r] = 0.0f;

    const float* arow[4];
#pragma unroll
    for (int t = 0; t < 4; ++t) {
        int mA = m0 + t * 16 + l16;
        if (mA >= M) mA = M - 1;                      // clamped load row
        arow[t] = A + (size_t)mA * lda;
    }

    // Packed B fragments for this N-tile: contiguous per lane, +512 elems/blk.
    const __bf16* bp = Wp + (((size_t)blockIdx.x * (K >> 5)) * 32 + lane) * 16;

    for (int kb = 0; kb < K; kb += 32) {   // K is always a multiple of 32 here
        U16bf b;
        b.v = *(const v16bf*)bp;
        bp += 512;

#pragma unroll
        for (int t = 0; t < 4; ++t) {
            const float* ap = arow[t] + kb + (half << 4);
            const float4 a0 = *(const float4*)(ap);
            const float4 a1 = *(const float4*)(ap + 4);
            const float4 a2 = *(const float4*)(ap + 8);
            const float4 a3 = *(const float4*)(ap + 12);
            U16bf a;
            a.e[0]  = (__bf16)a0.x; a.e[1]  = (__bf16)a0.y;
            a.e[2]  = (__bf16)a0.z; a.e[3]  = (__bf16)a0.w;
            a.e[4]  = (__bf16)a1.x; a.e[5]  = (__bf16)a1.y;
            a.e[6]  = (__bf16)a1.z; a.e[7]  = (__bf16)a1.w;
            a.e[8]  = (__bf16)a2.x; a.e[9]  = (__bf16)a2.y;
            a.e[10] = (__bf16)a2.z; a.e[11] = (__bf16)a2.w;
            a.e[12] = (__bf16)a3.x; a.e[13] = (__bf16)a3.y;
            a.e[14] = (__bf16)a3.z; a.e[15] = (__bf16)a3.w;
            acc[t].v = __builtin_amdgcn_wmma_f32_16x16x32_bf16(
                false, a.v, false, b.v, (short)0, acc[t].v, false, false);
        }
    }

    const int nB = n0 + l16;
    if (nB >= N) return;
    const float bval = (flags & GF_BIAS) ? bias[nB] : 0.0f;
#pragma unroll
    for (int t = 0; t < 4; ++t) {
#pragma unroll
        for (int r = 0; r < 8; ++r) {
            const int m = m0 + t * 16 + r + (half << 3);
            if (m < M) {
                float val = acc[t].e[r] + bval;
                if (flags & GF_RELU) val = fmaxf(val, 0.0f);
                if (flags & GF_RES)  val += res[(size_t)m * N + nB];
                C[(size_t)m * N + nB] = val;
            }
        }
    }
}

// ----------------------------------------------------------------------------
// Short attention (Lk <= 64): one block per batch; work item = (head, query).
// Two-pass softmax, 32-wide dot/acc fully unrolled (register resident).
// O[b,qi,h*32+j] = q[b,qi,h*32+j] + sum_k softmax(e)_k * v[k,h*32+j]
// ----------------------------------------------------------------------------
__global__ __launch_bounds__(128)
void attn_small_kernel(const float* __restrict__ q,
                       const float* __restrict__ k,
                       const float* __restrict__ v,
                       const unsigned char* __restrict__ mask,
                       float* __restrict__ O,
                       int Lq, int Lk, int sharedQ, int sharedK)
{
    const int b = blockIdx.x;
    const float scale = 0.08838834764831845f;  // 1/sqrt(128)

    for (int item = threadIdx.x; item < 4 * Lq; item += blockDim.x) {
        const int h  = item & 3;
        const int qi = item >> 2;
        const float* qr = q + (size_t)(sharedQ ? qi : (b * Lq + qi)) * 128 + h * 32;

        float mx = -1e30f;
        for (int kk = 0; kk < Lk; ++kk) {
            if (mask && mask[(size_t)b * Lk + kk]) continue;
            const float* kr = k + (size_t)(sharedK ? kk : (b * Lk + kk)) * 128 + h * 32;
            float d = 0.0f;
#pragma unroll
            for (int j = 0; j < 32; ++j) d += qr[j] * kr[j];
            mx = fmaxf(mx, d * scale);
        }

        float s = 0.0f;
        float av[32];
#pragma unroll
        for (int j = 0; j < 32; ++j) av[j] = 0.0f;

        for (int kk = 0; kk < Lk; ++kk) {
            if (mask && mask[(size_t)b * Lk + kk]) continue;
            const size_t kr_row = (size_t)(sharedK ? kk : (b * Lk + kk)) * 128 + h * 32;
            const float* kr = k + kr_row;
            float d = 0.0f;
#pragma unroll
            for (int j = 0; j < 32; ++j) d += qr[j] * kr[j];
            const float w = __expf(d * scale - mx);
            s += w;
            const float* vr = v + kr_row;
#pragma unroll
            for (int j = 0; j < 32; ++j) av[j] += w * vr[j];
        }

        const float inv = 1.0f / s;
        float* orow = O + (size_t)(b * Lq + qi) * 128 + h * 32;
#pragma unroll
        for (int j = 0; j < 32; ++j) orow[j] = qr[j] + av[j] * inv;
    }
}

// ----------------------------------------------------------------------------
// Long attention (Lq=1, Lk=6714, shared K/V): one block per (batch, head).
// Online softmax per thread over strided keys, then LDS tree combine of
// (max, sum, av[32]) partials.
// ----------------------------------------------------------------------------
__global__ __launch_bounds__(256)
void attn_big_kernel(const float* __restrict__ q,   // [B,128]
                     const float* __restrict__ k,   // [Lk,128] shared
                     const float* __restrict__ v,   // [Lk,128] shared
                     const unsigned char* __restrict__ mask, // [B,Lk]
                     float* __restrict__ O,         // [B,128]
                     int Lk)
{
    const int b = blockIdx.x >> 2;
    const int h = blockIdx.x & 3;
    const int t = threadIdx.x;
    const float scale = 0.08838834764831845f;

    const float* qr = q + (size_t)b * 128 + h * 32;
    float qreg[32];
#pragma unroll
    for (int j = 0; j < 32; ++j) qreg[j] = qr[j];

    float m = -1e30f, s = 0.0f;
    float av[32];
#pragma unroll
    for (int j = 0; j < 32; ++j) av[j] = 0.0f;

    for (int kk = t; kk < Lk; kk += 256) {
        if (mask[(size_t)b * Lk + kk]) continue;
        const float* kr = k + (size_t)kk * 128 + h * 32;
        float d = 0.0f;
#pragma unroll
        for (int j = 0; j < 32; ++j) d += qreg[j] * kr[j];
        d *= scale;
        if (d > m) {
            const float f = __expf(m - d);
            s *= f;
#pragma unroll
            for (int j = 0; j < 32; ++j) av[j] *= f;
            m = d;
        }
        const float w = __expf(d - m);
        s += w;
        const float* vr = v + (size_t)kk * 128 + h * 32;
#pragma unroll
        for (int j = 0; j < 32; ++j) av[j] += w * vr[j];
    }

    __shared__ float sm[256];
    __shared__ float ss[256];
    __shared__ float sav[32 * 256];
    sm[t] = m; ss[t] = s;
#pragma unroll
    for (int j = 0; j < 32; ++j) sav[j * 256 + t] = av[j];
    __syncthreads();

    for (int off = 128; off > 0; off >>= 1) {
        if (t < off) {
            const float m1 = sm[t], m2 = sm[t + off];
            const float mm = fmaxf(m1, m2);
            const float f1 = __expf(m1 - mm), f2 = __expf(m2 - mm);
            ss[t] = ss[t] * f1 + ss[t + off] * f2;
#pragma unroll
            for (int j = 0; j < 32; ++j)
                sav[j * 256 + t] = sav[j * 256 + t] * f1 + sav[j * 256 + t + off] * f2;
            sm[t] = mm;
        }
        __syncthreads();
    }

    if (t < 32)
        O[(size_t)b * 128 + h * 32 + t] = qr[t] + sav[t * 256] / ss[0];
}

// ----------------------------------------------------------------------------
// LayerNorm over last dim (=128): one block per row.
// ----------------------------------------------------------------------------
__global__ __launch_bounds__(128)
void layernorm_kernel(const float* __restrict__ X, const float* __restrict__ g,
                      const float* __restrict__ be, float* __restrict__ Y)
{
    const int row = blockIdx.x;
    const int t   = threadIdx.x;
    const float x = X[(size_t)row * 128 + t];

    __shared__ float red[128];
    red[t] = x; __syncthreads();
    for (int off = 64; off > 0; off >>= 1) {
        if (t < off) red[t] += red[t + off];
        __syncthreads();
    }
    const float mean = red[0] * (1.0f / 128.0f);
    __syncthreads();

    const float xm = x - mean;
    red[t] = xm * xm; __syncthreads();
    for (int off = 64; off > 0; off >>= 1) {
        if (t < off) red[t] += red[t + off];
        __syncthreads();
    }
    const float var = red[0] * (1.0f / 128.0f);
    Y[(size_t)row * 128 + t] = xm * rsqrtf(var + 1e-5f) * g[t] + be[t];
}

// ----------------------------------------------------------------------------
// Misc elementwise kernels
// ----------------------------------------------------------------------------
__global__ __launch_bounds__(256)
void gather_kernel(const int* __restrict__ x, const float* __restrict__ emb,
                   float* __restrict__ feat)
{
    const int r  = blockIdx.x;
    const int id = x[r];
    feat[(size_t)r * 256 + threadIdx.x] = emb[(size_t)id * 256 + threadIdx.x];
}

__global__ __launch_bounds__(256)
void padmask_kernel(const int* __restrict__ x, unsigned char* __restrict__ m,
                    int n, int pad)
{
    const int i = blockIdx.x * 256 + threadIdx.x;
    if (i < n) m[i] = (x[i] == pad) ? 1 : 0;
}

__global__ __launch_bounds__(256)
void maskout_kernel(const unsigned char* __restrict__ bin, float* __restrict__ out,
                    int n)
{
    const int i = blockIdx.x * 256 + threadIdx.x;
    if (i < n && bin[i]) out[i] = -__builtin_inff();
}

// ----------------------------------------------------------------------------
// Host-side orchestration
// ----------------------------------------------------------------------------
struct MabP {
    const float *Wq, *bq, *Wk, *bk, *Wv, *bv, *g0, *be0, *W1, *b1, *W2, *b2, *g1, *be1;
};

static MabP getMab(void* const* d, int base)
{
    MabP p;
    p.Wq  = (const float*)d[base + 0];  p.bq  = (const float*)d[base + 1];
    p.Wk  = (const float*)d[base + 2];  p.bk  = (const float*)d[base + 3];
    p.Wv  = (const float*)d[base + 4];  p.bv  = (const float*)d[base + 5];
    p.g0  = (const float*)d[base + 6];  p.be0 = (const float*)d[base + 7];
    p.W1  = (const float*)d[base + 8];  p.b1  = (const float*)d[base + 9];
    p.W2  = (const float*)d[base + 10]; p.b2  = (const float*)d[base + 11];
    p.g1  = (const float*)d[base + 12]; p.be1 = (const float*)d[base + 13];
    return p;
}

static void gemm(const float* A, int lda, const float* W, const float* bias,
                 const float* res, float* C, int M, int N, int K, int flags,
                 __bf16* packBuf, hipStream_t s)
{
    const int nT = (N + 15) / 16;
    dim3 pgrid(nT, K / 32);
    packW_kernel<<<pgrid, 32, 0, s>>>(W, packBuf, N);
    dim3 grid(nT, (M + 63) / 64);
    gemm_bf16_wmma<<<grid, 32, 0, s>>>(A, lda, packBuf, bias, res, C, M, N, K, flags);
}

static void run_mab(const MabP& p,
                    const float* Q, int lda_q, int dq, bool sharedQ,
                    const float* Kin, int dk, bool sharedK,
                    int B_, int Lq, int Lk,
                    const unsigned char* mask,
                    float* out,
                    float* bq, float* bk, float* bv, float* bO, float* bH,
                    __bf16* packBuf, hipStream_t s)
{
    const int rowsQ = sharedQ ? Lq : B_ * Lq;
    const int rowsK = sharedK ? Lk : B_ * Lk;

    gemm(Q,   lda_q, p.Wq, p.bq, nullptr, bq, rowsQ, 128, dq, GF_BIAS, packBuf, s);
    gemm(Kin, dk,    p.Wk, p.bk, nullptr, bk, rowsK, 128, dk, GF_BIAS, packBuf, s);
    gemm(Kin, dk,    p.Wv, p.bv, nullptr, bv, rowsK, 128, dk, GF_BIAS, packBuf, s);

    if (Lk > 64) {
        attn_big_kernel<<<B_ * 4, 256, 0, s>>>(bq, bk, bv, mask, bO, Lk);
    } else {
        attn_small_kernel<<<B_, 128, 0, s>>>(bq, bk, bv, mask, bO, Lq, Lk,
                                             sharedQ ? 1 : 0, sharedK ? 1 : 0);
    }

    layernorm_kernel<<<B_ * Lq, 128, 0, s>>>(bO, p.g0, p.be0, bO);
    gemm(bO, 128, p.W1, p.b1, nullptr, bH, B_ * Lq, 128, 128, GF_BIAS | GF_RELU,
         packBuf, s);
    gemm(bH, 128, p.W2, p.b2, bO,     bO, B_ * Lq, 128, 128, GF_BIAS | GF_RES,
         packBuf, s);
    layernorm_kernel<<<B_ * Lq, 128, 0, s>>>(bO, p.g1, p.be1, out);
}

extern "C" void kernel_launch(void* const* d_in, const int* in_sizes, int n_in,
                              void* d_out, int out_size, void* d_ws, size_t ws_size,
                              hipStream_t stream)
{
    (void)in_sizes; (void)n_in; (void)out_size; (void)ws_size;
    const int B_ = 1024, S_ = 64, DEMB = 256, DHID = 128;
    const int NITEMS = 6714, NINDS = 32;

    const int*           x     = (const int*)d_in[0];
    const unsigned char* bin_x = (const unsigned char*)d_in[1];
    const float*         emb   = (const float*)d_in[2];

    // ---- workspace carve-out (floats) --------------------------------------
    float* ws = (float*)d_ws;
    size_t off = 0;
    auto alloc = [&](size_t n) { float* p = ws + off; off += n; return p; };
    float* feat = alloc((size_t)B_ * S_ * DEMB);     // 16.8M
    float* code = alloc((size_t)B_ * S_ * DHID);     //  8.4M
    float* Hc   = alloc((size_t)B_ * NINDS * DHID);  //  4.2M
    float* bq   = alloc((size_t)B_ * S_ * DHID);
    float* bk   = alloc((size_t)B_ * S_ * DHID);
    float* bv   = alloc((size_t)B_ * S_ * DHID);
    float* bO   = alloc((size_t)B_ * S_ * DHID);
    float* bH   = alloc((size_t)B_ * S_ * DHID);
    float* sig  = alloc((size_t)B_ * 2 * DHID);
    float* sc   = alloc((size_t)B_ * DHID);
    float* sc2  = alloc((size_t)B_ * DHID);
    float* kE   = alloc((size_t)NITEMS * DHID);
    float* vE   = alloc((size_t)NITEMS * DHID);
    float* hcls = alloc((size_t)B_ * DHID);
    __bf16* packBuf = (__bf16*)(ws + off);           // 2M bf16 (covers ff2)
    off += (size_t)1048576;
    unsigned char* padmask = (unsigned char*)(ws + off);
    off += ((size_t)B_ * S_ + 3) / 4;

    // ---- embedding gather + pad mask --------------------------------------
    gather_kernel<<<B_ * S_, 256, 0, stream>>>(x, emb, feat);
    padmask_kernel<<<(B_ * S_ + 255) / 256, 256, 0, stream>>>(x, padmask, B_ * S_, NITEMS);

    // ---- encoder: 4 x ISAB --------------------------------------------------
    const float* codeIn = feat;
    int din = DEMB;
    for (int i = 0; i < 4; ++i) {
        const int base = 3 + 29 * i;
        const float* I = (const float*)d_in[base + 0];
        MabP m0 = getMab(d_in, base + 1);
        MabP m1 = getMab(d_in, base + 15);
        // mab0: Q = I (shared, Lq=32, dq=128); K = code (Lk=64, dk=din); masked
        run_mab(m0, I, 128, 128, true, codeIn, din, false,
                B_, NINDS, S_, padmask, Hc, bq, bk, bv, bO, bH, packBuf, stream);
        // mab1: Q = code (Lq=64, dq=din); K = Hc (Lk=32, dk=128); unmasked
        run_mab(m1, codeIn, din, din, false, Hc, 128, false,
                B_, S_, NINDS, nullptr, code, bq, bk, bv, bO, bH, packBuf, stream);
        codeIn = code;
        din = DHID;
    }

    // ---- PMA ---------------------------------------------------------------
    const float* Smat = (const float*)d_in[119];
    MabP pma = getMab(d_in, 120);
    run_mab(pma, Smat, 128, 128, true, code, 128, false,
            B_, 2, S_, padmask, sig, bq, bk, bv, bO, bH, packBuf, stream);

    // ---- dec1: 2 x SAB ------------------------------------------------------
    for (int i = 0; i < 2; ++i) {
        MabP d1 = getMab(d_in, 134 + 14 * i);
        run_mab(d1, sig, 128, 128, false, sig, 128, false,
                B_, 2, 2, nullptr, sig, bq, bk, bv, bO, bH, packBuf, stream);
    }

    // ---- ff1: logit_cls -----------------------------------------------------
    const float* f1W1 = (const float*)d_in[162];
    const float* f1b1 = (const float*)d_in[163];
    const float* f1W2 = (const float*)d_in[164];
    const float* f1b2 = (const float*)d_in[165];
    float* outF = (float*)d_out;
    // sig_cls = sig[:,0,:] (row stride 256)
    gemm(sig, 256, f1W1, f1b1, nullptr, hcls, B_, 128, 128, GF_BIAS | GF_RELU,
         packBuf, stream);
    gemm(hcls, 128, f1W2, f1b2, nullptr, outF, B_, 20, 128, GF_BIAS, packBuf, stream);

    // ---- dec2: 2 x MAB over embedding table (shared K, Lk=6714) -------------
    const float* scQ = sig + 128;   // sig_comp = sig[:,1,:] (row stride 256)
    int ldaQ = 256;
    for (int i = 0; i < 2; ++i) {
        MabP d2 = getMab(d_in, 166 + 14 * i);
        float* dst = (i == 0) ? sc : sc2;
        run_mab(d2, scQ, ldaQ, 128, false, emb, 256, true,
                B_, 1, NITEMS, bin_x, dst, bq, kE, vE, bO, bH, packBuf, stream);
        scQ = dst;
        ldaQ = 128;
    }

    // ---- ff2: logit_comp + -inf masking ------------------------------------
    const float* f2W = (const float*)d_in[194];
    const float* f2b = (const float*)d_in[195];
    float* logitComp = outF + (size_t)B_ * 20;
    gemm(sc2, 128, f2W, f2b, nullptr, logitComp, B_, NITEMS, 128, GF_BIAS,
         packBuf, stream);
    maskout_kernel<<<(B_ * NITEMS + 255) / 256, 256, 0, stream>>>(bin_x, logitComp,
                                                                  B_ * NITEMS);
}
